// Dense_EI_56779467653691
// MI455X (gfx1250) — matrix operand order repaired and checked by reading, not compiled, not run on hardware
//
#include <hip/hip_runtime.h>

typedef __attribute__((ext_vector_type(16))) __bf16 v16bf;
typedef __attribute__((ext_vector_type(8)))  __bf16 v8bf;
typedef __attribute__((ext_vector_type(8)))  float  v8f;

constexpr int Mdim = 8192;   // 4 * 2048
constexpr int Ndim = 4096;
constexpr int Kdim = 4096;
constexpr int NE   = 3277;   // round(4096 * 0.8)
constexpr float INH = -4.0f; // -P / (1 - P)

constexpr int BM = 128, BN = 128, BK = 32;
constexpr int SA = 40;       // LDS row stride (bf16) for A (m-major, BK cols)
constexpr int SB = 40;       // LDS row stride (bf16) for B (n-major, BK cols)
constexpr int KSTEPS = Kdim / BK;

union FragAB { v16bf v; v8bf h[2]; };

__global__ __launch_bounds__(256)
void dense_ei_wmma(const float* __restrict__ x,
                   const unsigned char* __restrict__ w,
                   const float* __restrict__ scale_p,
                   float* __restrict__ out)
{
  __shared__ __bf16 Al[2][BM * SA];
  __shared__ __bf16 Bl[2][BN * SB];

  const int tid = threadIdx.x;
  const int m0  = blockIdx.y * BM;
  const int n0  = blockIdx.x * BN;

  // global->reg staging assignments
  const int arow = tid >> 3;        // 0..31: row within 32-row pass
  const int acol = (tid & 7) * 4;   // k offset within tile (float4)
  const int bkg  = tid >> 5;        // 0..7 : k-group of 4 rows
  const int bng  = (tid & 31) * 4;  // n offset within tile (uchar4)

  float4       areg[4];
  unsigned int breg[4];

  auto load_global = [&](int k0) {
    #pragma unroll
    for (int p = 0; p < 4; ++p) {
      const float* src = x + (size_t)(m0 + p * 32 + arow) * Kdim + k0 + acol;
      areg[p] = *reinterpret_cast<const float4*>(src);
    }
    #pragma unroll
    for (int j = 0; j < 4; ++j) {
      const unsigned char* src = w + (size_t)(k0 + bkg * 4 + j) * Ndim + n0 + bng;
      breg[j] = *reinterpret_cast<const unsigned int*>(src);
    }
  };

  auto store_lds = [&](int buf, int k0) {
    // A: fuse EI column scale, convert f32 -> bf16
    #pragma unroll
    for (int p = 0; p < 4; ++p) {
      union { __bf16 h[4]; unsigned long long u; } pk;
      const float f[4] = {areg[p].x, areg[p].y, areg[p].z, areg[p].w};
      #pragma unroll
      for (int jj = 0; jj < 4; ++jj) {
        const int kg = k0 + acol + jj;
        const float fac = (kg < NE) ? 1.0f : INH;
        pk.h[jj] = (__bf16)(f[jj] * fac);
      }
      *reinterpret_cast<unsigned long long*>(
          &Al[buf][(p * 32 + arow) * SA + acol]) = pk.u;
    }
    // B: 4x4 micro-transpose to n-major; {0,1} -> bf16 via bit trick
    #pragma unroll
    for (int nn = 0; nn < 4; ++nn) {
      unsigned long long u = 0;
      #pragma unroll
      for (int j = 0; j < 4; ++j) {
        const unsigned long long bit = (breg[j] >> (8 * nn)) & 1u;
        u |= (bit * 0x3F80ull) << (16 * j);   // bf16(1.0) = 0x3F80
      }
      *reinterpret_cast<unsigned long long*>(
          &Bl[buf][(bng + nn) * SB + bkg * 4]) = u;
    }
  };

  const int lane = tid & 31;
  const int m16  = lane & 15;
  const int half = lane >> 4;
  const int wid  = tid >> 5;
  const int wm   = (wid & 3) * 32;   // wave row base within block tile
  const int wn   = (wid >> 2) * 64;  // wave col base within block tile

  v8f acc[2][4] = {};

  auto compute = [&](int buf) {
    // A fragment: 16-bit A 16x32 layout — elems 0..7 -> k=half*8+e,
    // elems 8..15 -> k=16+half*8+e  => two contiguous 16B LDS loads
    FragAB a[2], b[4];
    #pragma unroll
    for (int i = 0; i < 2; ++i) {
      const __bf16* ap = &Al[buf][(wm + i * 16 + m16) * SA + half * 8];
      a[i].h[0] = *reinterpret_cast<const v8bf*>(ap);
      a[i].h[1] = *reinterpret_cast<const v8bf*>(ap + 16);
    }
    // B fragment: lane holds column n=lane%16, elems 0..7 -> k=half*8+e, elems 8..15 -> k=16+half*8+e
    #pragma unroll
    for (int j = 0; j < 4; ++j) {
      const __bf16* bp = &Bl[buf][(wn + j * 16 + m16) * SB + half * 8];
      b[j].h[0] = *reinterpret_cast<const v8bf*>(bp);
      b[j].h[1] = *reinterpret_cast<const v8bf*>(bp + 16);
    }
    #pragma unroll
    for (int i = 0; i < 2; ++i)
      #pragma unroll
      for (int j = 0; j < 4; ++j)
        acc[i][j] = __builtin_amdgcn_wmma_f32_16x16x32_bf16(
            false, a[i].v, false, b[j].v, (short)0, acc[i][j], false, false);
  };

  // Software pipeline, 2x manual unroll over the double buffer.
  load_global(0);
  store_lds(0, 0);
  load_global(BK);

  #pragma unroll 1
  for (int kt = 0; kt < KSTEPS; kt += 2) {
    __syncthreads();                                     // lds[0] ready
    if (kt + 1 < KSTEPS) store_lds(1, (kt + 1) * BK);
    compute(0);
    if (kt + 2 < KSTEPS) load_global((kt + 2) * BK);

    __syncthreads();                                     // lds[1] ready
    if (kt + 2 < KSTEPS) store_lds(0, (kt + 2) * BK);
    if (kt + 1 < KSTEPS) compute(1);
    if (kt + 3 < KSTEPS) load_global((kt + 3) * BK);
  }

  // Epilogue: C/D f32 16x16 layout — lane n=lane%16, VGPR r -> row r+8*half
  const float s = scale_p[0];
  #pragma unroll
  for (int i = 0; i < 2; ++i) {
    #pragma unroll
    for (int j = 0; j < 4; ++j) {
      const int row = m0 + wm + i * 16 + half * 8;
      const int col = n0 + wn + j * 16 + m16;
      float* op = out + (size_t)row * Ndim + col;
      #pragma unroll
      for (int r = 0; r < 8; ++r)
        op[(size_t)r * Ndim] = acc[i][j][r] * s;
    }
  }
}

extern "C" void kernel_launch(void* const* d_in, const int* in_sizes, int n_in,
                              void* d_out, int out_size, void* d_ws, size_t ws_size,
                              hipStream_t stream) {
  const float*         x     = (const float*)d_in[0];
  const unsigned char* w     = (const unsigned char*)d_in[1];  // jnp bool: 1 byte
  const float*         scale = (const float*)d_in[2];
  float*               out   = (float*)d_out;

  dim3 grid(Ndim / BN, Mdim / BM);  // 32 x 64 = 2048 workgroups
  dense_ei_wmma<<<grid, dim3(256), 0, stream>>>(x, w, scale, out);
}
